// Multi_DefectModel_Graph2_22986664968800
// MI455X (gfx1250) — matrix operand order repaired and checked by reading, not compiled, not run on hardware
//
#include <hip/hip_runtime.h>
#include <hip/hip_bf16.h>

// ---------------- problem constants (match reference) ----------------
#define NN    10000   // nodes
#define NE    80000   // edges
#define NG    64      // graphs
#define NHEAD 4
#define DHEAD 512
#define HD    2048    // NHEAD*DHEAD
#define EMB   768
#define HF    512
#define SLOPE 0.2f

// ---------------- WMMA types ----------------
typedef __bf16 v16bf __attribute__((ext_vector_type(16)));
typedef float  v8f   __attribute__((ext_vector_type(8)));

union FragBF {
    v16bf v;
    unsigned int u[8];
};

__device__ __forceinline__ unsigned int f2bf_pk(float lo, float hi) {
    // pack two fp32 -> two bf16 (round-to-nearest-even) in one dword
    unsigned int ulo = __float_as_uint(lo);
    unsigned int uhi = __float_as_uint(hi);
    ulo = (ulo + 0x7FFFu + ((ulo >> 16) & 1u)) >> 16;
    uhi = (uhi + 0x7FFFu + ((uhi >> 16) & 1u)) & 0xFFFF0000u;
    return ulo | uhi;
}

// =====================================================================
// Tiled bf16 WMMA GEMM:  C[M,N] = act(A[M,K] @ B[K,N] + bias)
// 64(M) x 128(N) tile per 256-thread block; 8 waves in a 4x2 grid,
// each wave: one 16x32 A fragment reused against four 32x16 B fragments
// -> four 16x16 f32 accumulators (4 v_wmma per K-step per wave).
// N must be a multiple of 128 (true for all layers here: 2048, 512).
// Out-of-range A rows are index-clamped (their C rows are never stored),
// so staging is branchless.
// =====================================================================
__global__ __launch_bounds__(256) void gemm_bf16_wmma(
    const float* __restrict__ A, const float* __restrict__ B,
    const float* __restrict__ bias, float* __restrict__ C,
    int M, int N, int K, int act)
{
    __shared__ __align__(16) unsigned short Asl[64 * 34];   // [row][k]
    __shared__ __align__(16) unsigned short Bsl[128 * 34];  // [col][k]

    const int t    = threadIdx.x;
    const int lane = t & 31;
    const int w    = t >> 5;
    const int wr   = w & 3;     // wave M sub-tile (0..3), 16 rows
    const int wc   = w >> 2;    // wave N sub-tile (0..1), 64 cols
    const int bm   = blockIdx.x * 64;
    const int bn   = blockIdx.y * 128;

    v8f acc[4] = {v8f{}, v8f{}, v8f{}, v8f{}};

    // staging coordinates (hoisted out of the K loop)
    const int ar_  = (t >> 3);             // A row 0..31 (x2 iters -> 64)
    const int ac4  = (t & 7) << 2;         // A k-group-of-4
    const int bn_  = (t & 127);            // B col 0..127
    const int bk4  = (t >> 7) << 2;        // B k-group-of-4 (x4 iters -> 32)

    for (int kb = 0; kb < K; kb += 32) {
        // ---- stage A tile (64 x 32) via float4 loads, packed bf16 stores ----
        #pragma unroll
        for (int i = 0; i < 2; ++i) {
            int r   = ar_ + i * 32;
            int row = bm + r;
            row = row < M ? row : (M - 1);            // branchless clamp
            const float4 v = *(const float4*)&A[(size_t)row * K + kb + ac4];
            unsigned int* d = (unsigned int*)&Asl[r * 34 + ac4];
            d[0] = f2bf_pk(v.x, v.y);
            d[1] = f2bf_pk(v.z, v.w);
        }
        // ---- stage B tile (32 x 128) transposed: 4 k per thread, packed ----
        #pragma unroll
        for (int i = 0; i < 4; ++i) {
            int kk = bk4 + i * 8;
            const float* bp = &B[(size_t)(kb + kk) * N + bn + bn_];
            float v0 = bp[0];
            float v1 = bp[(size_t)N];
            float v2 = bp[(size_t)2 * N];
            float v3 = bp[(size_t)3 * N];
            unsigned int* d = (unsigned int*)&Bsl[bn_ * 34 + kk];
            d[0] = f2bf_pk(v0, v1);
            d[1] = f2bf_pk(v2, v3);
        }
        // prefetch next K tile while this one computes
        if (kb + 32 < K) {
            __builtin_prefetch(&A[(size_t)(bm + ar_) * K + kb + 32 + ac4], 0, 3);
            __builtin_prefetch(&B[(size_t)(kb + 32 + bk4) * N + bn + bn_], 0, 3);
        }
        __syncthreads();

        // ---- A fragment (16x32), ISA 16-bit A layout ----
        FragBF af;
        {
            int m = lane & 15, sel = lane >> 4;
            int rowoff = (wr * 16 + m) * 34;
            #pragma unroll
            for (int r = 0; r < 8; ++r) {
                int k0 = ((r >> 2) << 4) + (sel << 3) + ((r & 3) << 1);
                af.u[r] = *(const unsigned int*)&Asl[rowoff + k0];
            }
        }
        // ---- four B fragments (32x16 each), ISA 16-bit B layout ----
        FragBF bfr[4];
        {
            int n = lane & 15, sel = lane >> 4;
            #pragma unroll
            for (int a = 0; a < 4; ++a) {
                int coff = (wc * 64 + a * 16 + n) * 34;
                #pragma unroll
                for (int r = 0; r < 8; ++r) {
                    int k0 = ((r >> 2) << 4) + (sel << 3) + ((r & 3) << 1);
                    bfr[a].u[r] = *(const unsigned int*)&Bsl[coff + k0];
                }
            }
        }

        #pragma unroll
        for (int a = 0; a < 4; ++a) {
            acc[a] = __builtin_amdgcn_wmma_f32_16x16x32_bf16(
                false, af.v, false, bfr[a].v, (short)0, acc[a], false, false);
        }
        __syncthreads();
    }

    // ---- store with fused bias + ELU; C/D layout: VGPR v -> rows v, v+8 ----
    const int mrow0 = bm + wr * 16 + ((lane >> 4) << 3);
    const int ncol0 = bn + wc * 64 + (lane & 15);
    #pragma unroll
    for (int vv = 0; vv < 8; ++vv) {
        int row = mrow0 + vv;
        if (row < M) {
            #pragma unroll
            for (int a = 0; a < 4; ++a) {
                int col = ncol0 + a * 16;
                float val = acc[a][vv] + (bias ? bias[col] : 0.0f);
                if (act) val = val > 0.0f ? val : (__expf(val) - 1.0f);
                C[(size_t)row * N + col] = val;
            }
        }
    }
}

// =====================================================================
// Attention helpers
// =====================================================================

// one block per node, one wave per head: el/er dot products over DHEAD
__global__ __launch_bounds__(128) void k_el_er(
    const float* __restrict__ feat, const float* __restrict__ al,
    const float* __restrict__ ar, float* __restrict__ el,
    float* __restrict__ er, int nNodes)
{
    int n = blockIdx.x;
    if (n >= nNodes) return;
    int h = threadIdx.x >> 5;
    int lane = threadIdx.x & 31;
    const float* f  = feat + (size_t)n * HD + h * DHEAD;
    const float* a1 = al + h * DHEAD;
    const float* a2 = ar + h * DHEAD;
    float s1 = 0.0f, s2 = 0.0f;
    for (int d = lane; d < DHEAD; d += 32) {
        float fv = f[d];
        s1 += fv * a1[d];
        s2 += fv * a2[d];
    }
    for (int o = 16; o; o >>= 1) {
        s1 += __shfl_xor(s1, o, 32);
        s2 += __shfl_xor(s2, o, 32);
    }
    if (lane == 0) { el[n * 4 + h] = s1; er[n * 4 + h] = s2; }
}

__global__ void k_fill_f(float* p, int n, float v) {
    int i = blockIdx.x * blockDim.x + threadIdx.x;
    if (i < n) p[i] = v;
}
__global__ void k_fill_i(int* p, int n, int v) {
    int i = blockIdx.x * blockDim.x + threadIdx.x;
    if (i < n) p[i] = v;
}
// out[row, j] = b[j]   (initializes aggregation buffer with the GAT bias)
__global__ void k_bias_init(float* __restrict__ out, const float* __restrict__ b,
                            int total, int cols) {
    int i = blockIdx.x * blockDim.x + threadIdx.x;
    if (i < total) out[i] = b[i & (cols - 1)];   // cols is a power of two (2048)
}

// leaky_relu(el[src]+er[dst]) + segment-max via monotonic int-encoded atomicMax
__global__ void k_edge_logit(const int* __restrict__ src, const int* __restrict__ dst,
                             const float* __restrict__ el, const float* __restrict__ er,
                             float* __restrict__ ebuf, int* __restrict__ emax, int nE)
{
    int i = blockIdx.x * blockDim.x + threadIdx.x;
    if (i >= nE * 4) return;
    int e = i >> 2, h = i & 3;
    int s = src[e], d = dst[e];
    float x = el[s * 4 + h] + er[d * 4 + h];
    x = x > 0.0f ? x : SLOPE * x;
    ebuf[i] = x;
    int enc = __float_as_int(x);
    enc = enc >= 0 ? enc : (enc ^ 0x7FFFFFFF);   // order-preserving float->int
    atomicMax(&emax[d * 4 + h], enc);
}

// ex = exp(e - emax[dst]); den[dst] += ex
__global__ void k_edge_exp(const int* __restrict__ dst, float* __restrict__ ebuf,
                           const int* __restrict__ emax, float* __restrict__ den, int nE)
{
    int i = blockIdx.x * blockDim.x + threadIdx.x;
    if (i >= nE * 4) return;
    int e = i >> 2, h = i & 3;
    int d = dst[e];
    int enc = emax[d * 4 + h];
    enc = enc >= 0 ? enc : (enc ^ 0x7FFFFFFF);
    float mx = __int_as_float(enc);
    float ex = __expf(ebuf[i] - mx);
    ebuf[i] = ex;
    atomicAdd(&den[d * 4 + h], ex);
}

// out[dst] += feat[src] * a  (block per edge, 2048 floats)
__global__ __launch_bounds__(256) void k_scatter(
    const int* __restrict__ src, const int* __restrict__ dst,
    const float* __restrict__ feat, const float* __restrict__ ex,
    const float* __restrict__ den, float* __restrict__ out, int nE)
{
    int e = blockIdx.x;
    if (e >= nE) return;
    int s = src[e], d = dst[e];
    const float* fs = feat + (size_t)s * HD;
    float* od = out + (size_t)d * HD;
    #pragma unroll
    for (int i = 0; i < 8; ++i) {
        int idx = threadIdx.x + (i << 8);
        int h = idx >> 9;
        float a = ex[e * 4 + h] / den[d * 4 + h];
        atomicAdd(&od[idx], fs[idx] * a);
    }
}

// =====================================================================
// Pooling / head
// =====================================================================
__global__ __launch_bounds__(256) void k_pool(
    const float* __restrict__ h, const int* __restrict__ gid,
    float* __restrict__ hg, float* __restrict__ counts, int nNodes)
{
    int n = blockIdx.x;
    if (n >= nNodes) return;
    int g = gid[n];
    if (threadIdx.x == 0) atomicAdd(&counts[g], 1.0f);
    #pragma unroll
    for (int i = 0; i < 2; ++i) {
        int idx = threadIdx.x + (i << 8);
        atomicAdd(&hg[(size_t)g * HF + idx], h[(size_t)n * HF + idx]);
    }
}

__global__ void k_bn(const float* __restrict__ hg, const float* __restrict__ counts,
                     const float* __restrict__ gamma, const float* __restrict__ beta,
                     float* __restrict__ outp)
{
    int i = blockIdx.x * blockDim.x + threadIdx.x;
    if (i >= NG * HF) return;
    int g = i >> 9, j = i & 511;
    float c = counts[g];
    c = c > 1.0f ? c : 1.0f;
    float v = hg[i] / c;
    v = v * rsqrtf(1.0f + 1e-5f) * gamma[j] + beta[j];
    outp[i] = v;
}

// logits: (64 x 512) @ (512 x 2) + bias; one wave per (graph, class)
__global__ __launch_bounds__(32) void k_final(
    const float* __restrict__ hg, const float* __restrict__ Wf,
    const float* __restrict__ bfv, float* __restrict__ out)
{
    int g = blockIdx.x, c = blockIdx.y;
    int lane = threadIdx.x;
    float s = 0.0f;
    for (int d = lane; d < HF; d += 32) s += hg[g * HF + d] * Wf[d * 2 + c];
    for (int o = 16; o; o >>= 1) s += __shfl_xor(s, o, 32);
    if (lane == 0) out[g * 2 + c] = s + bfv[c];
}

// =====================================================================
// Host launcher
// =====================================================================
extern "C" void kernel_launch(void* const* d_in, const int* in_sizes, int n_in,
                              void* d_out, int out_size, void* d_ws, size_t ws_size,
                              hipStream_t stream)
{
    (void)in_sizes; (void)n_in; (void)out_size; (void)ws_size;

    const float* node_feat = (const float*)d_in[0];
    // d_in[1] func_feat: dead branch (h_func never reaches the output)
    const float* W1    = (const float*)d_in[2];
    const float* al1   = (const float*)d_in[3];
    const float* ar1   = (const float*)d_in[4];
    const float* b1    = (const float*)d_in[5];
    const float* W2    = (const float*)d_in[6];
    const float* al2   = (const float*)d_in[7];
    const float* ar2   = (const float*)d_in[8];
    const float* b2    = (const float*)d_in[9];
    const float* Wfc   = (const float*)d_in[10];
    const float* bfc   = (const float*)d_in[11];
    const float* Wh    = (const float*)d_in[14];
    const float* bh    = (const float*)d_in[15];
    const float* gamma = (const float*)d_in[16];
    const float* beta  = (const float*)d_in[17];
    const float* Whfc  = (const float*)d_in[18];
    const float* bhfc  = (const float*)d_in[19];
    const float* Wf    = (const float*)d_in[20];
    const float* bfv   = (const float*)d_in[21];
    const int*   srcI  = (const int*)d_in[22];
    const int*   dstI  = (const int*)d_in[23];
    const int*   gid   = (const int*)d_in[24];
    float* outp = (float*)d_out;

    // ---- workspace carve-up (floats) ----
    float* ws = (float*)d_ws;
    size_t o = 0;
    float* bufA   = ws + o; o += (size_t)NN * HD;   // activations ping
    float* bufB   = ws + o; o += (size_t)NN * HD;   // activations pong
    float* el     = ws + o; o += (size_t)NN * 4;
    float* er     = ws + o; o += (size_t)NN * 4;
    int*   emax   = (int*)(ws + o); o += (size_t)NN * 4;
    float* den    = ws + o; o += (size_t)NN * 4;
    float* ebuf   = ws + o; o += (size_t)NE * 4;
    float* hg     = ws + o; o += (size_t)NG * HF;
    float* hgb    = ws + o; o += (size_t)NG * HF;
    float* counts = ws + o; o += NG;

    auto gemm = [&](const float* A, const float* B, const float* bias, float* C,
                    int M, int N, int K, int act) {
        dim3 grid((M + 63) / 64, N / 128);
        gemm_bf16_wmma<<<grid, 256, 0, stream>>>(A, B, bias, C, M, N, K, act);
    };
    const int EH  = NE * 4;
    const int NH4 = NN * 4;

    auto gat_attention = [&](const float* feat, const float* al, const float* ar,
                             const float* b, float* out) {
        k_el_er<<<NN, 128, 0, stream>>>(feat, al, ar, el, er, NN);
        k_fill_i<<<(NH4 + 255) / 256, 256, 0, stream>>>(emax, NH4, (int)0x80000000);
        k_fill_f<<<(NH4 + 255) / 256, 256, 0, stream>>>(den, NH4, 0.0f);
        k_edge_logit<<<(EH + 255) / 256, 256, 0, stream>>>(srcI, dstI, el, er, ebuf, emax, NE);
        k_edge_exp<<<(EH + 255) / 256, 256, 0, stream>>>(dstI, ebuf, emax, den, NE);
        k_bias_init<<<(NN * HD + 255) / 256, 256, 0, stream>>>(out, b, NN * HD, HD);
        k_scatter<<<NE, 256, 0, stream>>>(srcI, dstI, feat, ebuf, den, out, NE);
    };

    // ---- GAT layer 1: feat1 = node_feat @ W1; aggregate -> bufB ----
    gemm(node_feat, W1, nullptr, bufA, NN, HD, EMB, 0);
    gat_attention(bufA, al1, ar1, b1, bufB);

    // ---- GAT layer 2: feat2 = h1 @ W2; aggregate -> bufB ----
    gemm(bufB, W2, nullptr, bufA, NN, HD, HD, 0);
    gat_attention(bufA, al2, ar2, b2, bufB);

    // ---- h = elu(h2 @ Wfc + bfc) ----
    gemm(bufB, Wfc, bfc, bufA, NN, HF, HD, 1);

    // ---- 8 hidden layers: h = elu(h @ Wh[i] + bh[i]) ----
    float* cur = bufA;
    float* nxt = bufB;
    for (int i = 0; i < 8; ++i) {
        gemm(cur, Wh + (size_t)i * HF * HF, bh + (size_t)i * HF, nxt, NN, HF, HF, 1);
        float* tmp = cur; cur = nxt; nxt = tmp;
    }
    // final node features are in `cur` (== bufA after 8 swaps)

    // ---- mean pooling per graph ----
    k_fill_f<<<(NG * HF + 255) / 256, 256, 0, stream>>>(hg, NG * HF, 0.0f);
    k_fill_f<<<1, 64, 0, stream>>>(counts, NG, 0.0f);
    k_pool<<<NN, 256, 0, stream>>>(cur, gid, hg, counts, NN);

    // ---- batchnorm (eval, mean=0, var=1) ----
    k_bn<<<(NG * HF + 255) / 256, 256, 0, stream>>>(hg, counts, gamma, beta, hgb);

    // ---- hg = elu(hg @ Whfc + bhfc) ----
    gemm(hgb, Whfc, bhfc, hg, NG, HF, HF, 1);

    // ---- logits = hg @ Wf + bf ----
    k_final<<<dim3(NG, 2), 32, 0, stream>>>(hg, Wf, bfv, outp);
}
